// WindowAttention_7799660609921
// MI455X (gfx1250) — matrix operand order repaired and checked by reading, not compiled, not run on hardware
//
#include <hip/hip_runtime.h>

typedef __attribute__((ext_vector_type(16))) __bf16 v16bf;
typedef __attribute__((ext_vector_type(8)))  float  v8f;

#define NTOK 49
#define NPAD 64   // zero-padded token count
#define CDIM 192
#define HEADS 6
#define HD 32
#define SSTR 50   // row stride (elements) for score / P matrices

// LDS byte offsets inside one arena (phase-aliased)
#define SX_OFF   0        // bf16 [64][192] staged input, zero-padded (phase1)
#define SS_OFF   0        // f32 [6][49][50] scores (phase2+); aliases sX
#define SQ_OFF   58800    // bf16 [6][64][32] (rows 49..63 zero)
#define SK_OFF   83376    // bf16 [6][64][32] (rows 49..63 zero)
#define SV_OFF   107952   // bf16 [6][32][64] V transposed (cols 49..63 zero)
#define SP_OFF   132528   // bf16 [6][49][50] softmax probs
#define AO_OFF   58800    // bf16 [64][192] attn out; aliases dead sQ (phase3)
#define SMEM_BYTES 161936

// workspace layout (bf16 weights)
#define QKVW_ELTS (3 * CDIM * CDIM)   // 110592
#define PROJW_ELTS (CDIM * CDIM)      // 36864
#define PROJW_WS_OFF (QKVW_ELTS * 2)  // bytes

union Frag {
  v16bf v;
  __bf16 h[16];
  unsigned int u[8];
};

union Pack8 {
  __bf16 hh[8];
  unsigned int uu[4];
};

__device__ __forceinline__ v8f wmma_bf16(const Frag& A, const Frag& B, v8f c) {
  return __builtin_amdgcn_wmma_f32_16x16x32_bf16(
      /*neg_a=*/false, A.v, /*neg_b=*/false, B.v,
      /*c_mod=*/(short)0, c, /*reuse_a=*/false, /*reuse_b=*/false);
}

// A-fragment (16x32 bf16): lane holds row (lane&15); VGPR v -> k pair at
// kb + {2v | 16+2(v-4)} + 8*(lane>>4). Two contiguous 16B runs.
__device__ __forceinline__ void load_afrag(Frag& A, const __bf16* rowBase,
                                           int kb, int halfL) {
  const unsigned int* pr = (const unsigned int*)rowBase;
  #pragma unroll
  for (int v = 0; v < 8; ++v) {
    const int k0 = kb + ((v < 4) ? 2 * v : 16 + 2 * (v - 4)) + 8 * halfL;
    A.u[v] = pr[k0 >> 1];
  }
}

// B-fragment (32x16 bf16): lane holds col n; k at kb + 8*halfL + {0..7, 16..23}.
__device__ __forceinline__ void load_bfrag_contig(Frag& B, const __bf16* colBase,
                                                  int kb, int halfL) {
  const unsigned int* p = (const unsigned int*)(colBase + kb + 8 * halfL);
  #pragma unroll
  for (int v = 0; v < 8; ++v) B.u[v] = p[(v < 4) ? v : v + 4];
}

__global__ __launch_bounds__(256)
void cvt_weights_kernel(const float* __restrict__ qkv_w,
                        const float* __restrict__ proj_w,
                        __bf16* __restrict__ qkv_wb,
                        __bf16* __restrict__ proj_wb) {
  const int i = blockIdx.x * blockDim.x + threadIdx.x;
  if (i < QKVW_ELTS) qkv_wb[i] = (__bf16)qkv_w[i];
  if (i < PROJW_ELTS) proj_wb[i] = (__bf16)proj_w[i];
}

__global__ __launch_bounds__(256)
void win_attn_kernel(const float* __restrict__ x,
                     const __bf16* __restrict__ qkv_wb,
                     const float* __restrict__ qkv_b,
                     const __bf16* __restrict__ proj_wb,
                     const float* __restrict__ proj_b,
                     const float* __restrict__ bias_table,
                     const int*   __restrict__ rel_index,
                     float* __restrict__ out)
{
  __shared__ __align__(16) char smem[SMEM_BYTES];
  __bf16* sX  = (__bf16*)(smem + SX_OFF);   // [64][192] zero-padded
  float*  sS  = (float*) (smem + SS_OFF);   // [6][49][50]
  __bf16* sQ  = (__bf16*)(smem + SQ_OFF);   // [6][64][32]
  __bf16* sK  = (__bf16*)(smem + SK_OFF);   // [6][64][32]
  __bf16* sVt = (__bf16*)(smem + SV_OFF);   // [6][32][64]  V^T
  __bf16* sP  = (__bf16*)(smem + SP_OFF);   // [6][49][50]
  __bf16* sO  = (__bf16*)(smem + AO_OFF);   // [64][192]

  const int tid   = threadIdx.x;
  const int lane  = tid & 31;           // wave32
  const int wave  = tid >> 5;           // 8 waves
  const int nIdx  = lane & 15;
  const int halfL = lane >> 4;
  const int b     = blockIdx.x;

  // ---------- phase 0: stage x[b] as bf16 (zero-pad rows 49..63) ----------
  const float* xb = x + (size_t)b * NTOK * CDIM;
  for (int i = tid; i < NPAD * CDIM; i += 256)
    sX[i] = (i < NTOK * CDIM) ? (__bf16)xb[i] : (__bf16)0.0f;
  // zero the pad rows of per-head q/k (rows 49..63) and v^T (cols 49..63)
  for (int i = tid; i < HEADS * (NPAD - NTOK) * HD; i += 256) {
    const int h  = i / ((NPAD - NTOK) * HD);
    const int rr = (i / HD) % (NPAD - NTOK);
    const int d  = i % HD;
    sQ[(h * NPAD + NTOK + rr) * HD + d] = (__bf16)0.0f;
    sK[(h * NPAD + NTOK + rr) * HD + d] = (__bf16)0.0f;
    sVt[(h * HD + d) * NPAD + NTOK + rr] = (__bf16)0.0f;
  }
  __syncthreads();

  const float qscale = 0.17677669529663687f;  // 1/sqrt(32)

  // ---------- phase 1a: Q,K via transposed GEMM  D[outCh][token] = W @ X^T ---
  // 24 M-tiles (384 q|k channels) x 4 N-tiles (tokens); nt invariant per wave
  // so the staged-X B-fragments stay VGPR-resident while weights stream.
  for (int t = wave; t < 96; t += 8) {
    const int nt = t & 3;                    // token tile (invariant per wave)
    const int mt = t >> 2;                   // 0..23 over q|k channels
    const int token  = nt * 16 + nIdx;
    const int outRow = mt * 16 + nIdx;       // A row = output channel
    v8f acc;
    {
      const float* bb = qkv_b + mt * 16 + 8 * halfL;   // bias along M
      #pragma unroll
      for (int r = 0; r < 8; ++r) acc[r] = bb[r];
    }
    #pragma unroll
    for (int kk = 0; kk < 6; ++kk) {
      const int kb = kk * 32;
      Frag A, B;
      load_afrag(A, qkv_wb + (size_t)outRow * CDIM, kb, halfL);
      load_bfrag_contig(B, sX + token * CDIM, kb, halfL);
      acc = wmma_bf16(A, B, acc);
    }
    const int outBase = mt * 16;
    const int which = __builtin_amdgcn_readfirstlane(outBase / CDIM); // 0=q 1=k
    const int rem   = outBase - which * CDIM;
    const int head  = __builtin_amdgcn_readfirstlane(rem >> 5);
    const int dBase = __builtin_amdgcn_readfirstlane(rem & 31) + 8 * halfL;
    const float mult = (which == 0) ? qscale : 1.0f;
    if (token < NTOK) {                       // one predicated 16B store
      Pack8 pk;
      #pragma unroll
      for (int r = 0; r < 8; ++r) pk.hh[r] = (__bf16)(acc[r] * mult);
      unsigned int* dw = (unsigned int*)(((which == 0) ? sQ : sK) +
                                         (head * NPAD + token) * HD + dBase);
      #pragma unroll
      for (int w4 = 0; w4 < 4; ++w4) dw[w4] = pk.uu[w4];
    }
  }

  // ---------- phase 1b: V = X @ Wv^T (original orientation, packed V^T store)
  for (int t = wave; t < 48; t += 8) {
    const int mt = t & 3, nt = t >> 2;       // mt invariant -> X A-frags resident
    const int col = 2 * CDIM + nt * 16 + nIdx;   // v block: 384..575
    v8f acc = {};
    {
      const float bias = qkv_b[col];
      #pragma unroll
      for (int r = 0; r < 8; ++r) acc[r] = bias;
    }
    const int aRow = mt * 16 + nIdx;
    #pragma unroll
    for (int kk = 0; kk < 6; ++kk) {
      const int kb = kk * 32;
      Frag A, B;
      load_afrag(A, sX + aRow * CDIM, kb, halfL);
      load_bfrag_contig(B, qkv_wb + (size_t)col * CDIM, kb, halfL);
      acc = wmma_bf16(A, B, acc);
    }
    const int head  = __builtin_amdgcn_readfirstlane(nt >> 1);
    const int dBase = __builtin_amdgcn_readfirstlane((nt & 1) * 16);
    const int d     = dBase + nIdx;
    // lane's 8 rows are consecutive in V^T -> one aligned 16B store,
    // branchless zeroing keeps the pad region zero.
    Pack8 pk;
    const int rowBase = mt * 16 + 8 * halfL;
    #pragma unroll
    for (int r = 0; r < 8; ++r)
      pk.hh[r] = (__bf16)((rowBase + r < NTOK) ? acc[r] : 0.0f);
    unsigned int* dw = (unsigned int*)(sVt + (head * HD + d) * NPAD + rowBase);
    #pragma unroll
    for (int w4 = 0; w4 < 4; ++w4) dw[w4] = pk.uu[w4];
  }
  __syncthreads();

  // ---------- phase 2: S = Q @ K^T (96 tiles across 8 waves) ----------
  for (int t = wave; t < HEADS * 16; t += 8) {
    const int h = __builtin_amdgcn_readfirstlane(t >> 4);
    const int tt = t & 15;
    const int mt = tt >> 2, nt = tt & 3;
    const int aRow = mt * 16 + nIdx;
    const int bTok = nt * 16 + nIdx;
    Frag A, B;
    load_afrag(A, sQ + (h * NPAD + aRow) * HD, 0, halfL);
    load_bfrag_contig(B, sK + (h * NPAD + bTok) * HD, 0, halfL);
    v8f acc = {};
    acc = wmma_bf16(A, B, acc);
    const int col = nt * 16 + nIdx;
    if (col < NTOK) {
      #pragma unroll
      for (int r = 0; r < 8; ++r) {
        const int row = mt * 16 + r + 8 * halfL;
        if (row < NTOK) sS[(h * NTOK + row) * SSTR + col] = acc[r];
      }
    }
  }
  __syncthreads();

  // ---------- phase 2b: softmax(S + rel_bias) -> P (bf16) ----------
  for (int r = tid; r < HEADS * NTOK; r += 256) {
    const int h = r / NTOK, m = r - h * NTOK;
    float* srow = sS + (h * NTOK + m) * SSTR;
    const int* ri = rel_index + m * NTOK;
    float mx = -1e30f;
    for (int j = 0; j < NTOK; ++j) {
      const float v = srow[j] + bias_table[ri[j] * HEADS + h];
      srow[j] = v;
      mx = fmaxf(mx, v);
    }
    float sum = 0.f;
    for (int j = 0; j < NTOK; ++j) {
      const float e = __expf(srow[j] - mx);
      srow[j] = e;
      sum += e;
    }
    const float rec = 1.f / sum;
    __bf16* prow = sP + (h * NTOK + m) * SSTR;
    for (int j = 0; j < NTOK; ++j)
      prow[j] = (__bf16)(srow[j] * rec);
  }
  __syncthreads();

  // ---------- phase 3a: O = P @ V (48 tiles across 8 waves) ----------
  for (int t = wave; t < HEADS * 8; t += 8) {
    const int h = __builtin_amdgcn_readfirstlane(t >> 3);
    const int tt = t & 7;
    const int mt = tt >> 1, nt = tt & 1;
    const int aRow = mt * 16 + nIdx;
    const int d = nt * 16 + nIdx;
    v8f acc = {};
    #pragma unroll
    for (int kk = 0; kk < 2; ++kk) {
      const int kb = kk * 32;
      Frag A, B;
      // A from P: per-element, predicated (rows/cols beyond 48 -> 0)
      const __bf16* pa = sP + (h * NTOK + (aRow < NTOK ? aRow : 0)) * SSTR;
      #pragma unroll
      for (int v = 0; v < 8; ++v) {
        const int k0 = kb + ((v < 4) ? 2 * v : 16 + 2 * (v - 4)) + 8 * halfL;
        A.h[2 * v]     = (aRow < NTOK && k0 < NTOK)     ? pa[k0]     : (__bf16)0.0f;
        A.h[2 * v + 1] = (aRow < NTOK && k0 + 1 < NTOK) ? pa[k0 + 1] : (__bf16)0.0f;
      }
      // B from V^T: element (k=token, n=dim) = sVt[h][d][token]; contiguous, zero-padded
      load_bfrag_contig(B, sVt + (h * HD + d) * NPAD, kb, halfL);
      acc = wmma_bf16(A, B, acc);
    }
    #pragma unroll
    for (int r = 0; r < 8; ++r) {
      const int row = mt * 16 + r + 8 * halfL;   // pad rows hold zeros (A rows pred to 0)
      sO[row * CDIM + h * HD + d] = (__bf16)acc[r];
    }
  }
  __syncthreads();

  // ---------- phase 3b: out^T-oriented proj  D[outCh][token] = Wp @ O^T -----
  // 12 M-tiles (192 out channels) x 4 N-tiles (tokens); nt invariant per wave
  // so the sO B-fragments stay resident; output store is 8 consecutive floats.
  float* ob = out + (size_t)b * NTOK * CDIM;
  for (int t = wave; t < 48; t += 8) {
    const int nt = t & 3;                    // token tile (invariant per wave)
    const int mt = t >> 2;                   // 0..11 over out channels
    const int token  = nt * 16 + nIdx;
    const int outRow = mt * 16 + nIdx;
    v8f acc;
    {
      const float* bb = proj_b + mt * 16 + 8 * halfL;
      #pragma unroll
      for (int r = 0; r < 8; ++r) acc[r] = bb[r];
    }
    #pragma unroll
    for (int kk = 0; kk < 6; ++kk) {
      const int kb = kk * 32;
      Frag A, B;
      load_afrag(A, proj_wb + (size_t)outRow * CDIM, kb, halfL);
      load_bfrag_contig(B, sO + token * CDIM, kb, halfL);
      acc = wmma_bf16(A, B, acc);
    }
    if (token < NTOK) {
      float* po = ob + (size_t)token * CDIM + mt * 16 + 8 * halfL;
      #pragma unroll
      for (int r = 0; r < 8; ++r) po[r] = acc[r];   // 2x global_store_b128
    }
  }
}

extern "C" void kernel_launch(void* const* d_in, const int* in_sizes, int n_in,
                              void* d_out, int out_size, void* d_ws, size_t ws_size,
                              hipStream_t stream) {
  const float* x          = (const float*)d_in[0];
  const float* qkv_w      = (const float*)d_in[1];
  const float* qkv_b      = (const float*)d_in[2];
  const float* proj_w     = (const float*)d_in[3];
  const float* proj_b     = (const float*)d_in[4];
  const float* bias_table = (const float*)d_in[5];
  const int*   rel_index  = (const int*)d_in[6];
  float* out = (float*)d_out;

  __bf16* qkv_wb  = (__bf16*)d_ws;
  __bf16* proj_wb = (__bf16*)((char*)d_ws + PROJW_WS_OFF);

  // 1) convert weights to bf16 once per launch (stream-ordered)
  {
    const int n = QKVW_ELTS;  // >= PROJW_ELTS
    cvt_weights_kernel<<<(n + 255) / 256, 256, 0, stream>>>(qkv_w, proj_w,
                                                            qkv_wb, proj_wb);
  }

  // 2) fused window attention, one workgroup per window
  const int B = in_sizes[0] / (NTOK * CDIM);  // 4096
  win_attn_kernel<<<B, 256, 0, stream>>>(x, qkv_wb, qkv_b, proj_wb, proj_b,
                                         bias_table, rel_index, out);
}
